// RegionProposeNetwork_56367150792707
// MI455X (gfx1250) — matrix operand order repaired and checked by reading, not compiled, not run on hardware
//
#include <hip/hip_runtime.h>

typedef __attribute__((ext_vector_type(16))) __bf16 v16bf;
typedef __attribute__((ext_vector_type(8)))  float  v8f;

static constexpr int BB    = 16;     // batch
static constexpr int FEAT  = 512;
static constexpr int OUTW  = 50;
static constexpr int PADW  = 52;     // padded spatial (1-px zero halo)
static constexpr int NPOS  = 2500;   // 50*50
static constexpr int AA    = 9;
static constexpr int NANCH = 22500;  // NPOS*AA
static constexpr int KTOT  = 4608;   // 9*512
static constexpr int KB_N  = 144;    // k-steps of 32

// ------------------------------------------------------------ async helpers

__device__ __forceinline__ void async_ld128(void* lds, const void* g) {
  // GLOBAL_LOAD_ASYNC_TO_LDS_B128: vdst = LDS offset (low 32 bits of flat LDS
  // pointer, per ISA flat->LDS aperture mapping), vaddr = 64-bit global addr.
  asm volatile("global_load_async_to_lds_b128 %0, %1, off"
               :: "v"((unsigned)(unsigned long long)lds), "v"(g)
               : "memory");
}
template <int N>
__device__ __forceinline__ void wait_async() {
  asm volatile("s_wait_asynccnt %0" :: "i"(N) : "memory");
}

// ---------------------------------------------------------------- prep kernels

// features NCHW f32 -> padded NHWC bf16  [b][52][52][512], zero halo
__global__ void k_cvt_feat_pad(const float* __restrict__ f, __bf16* __restrict__ x) {
  long i = (long)blockIdx.x * 256 + threadIdx.x;
  if (i >= (long)BB * PADW * PADW * FEAT) return;
  int c = (int)(i % FEAT);
  long t = i / FEAT;
  int pw = (int)(t % PADW); t /= PADW;
  int ph = (int)(t % PADW);
  int b  = (int)(t / PADW);
  float v = 0.f;
  if (ph >= 1 && ph <= OUTW && pw >= 1 && pw <= OUTW)
    v = f[((long)b * FEAT + c) * NPOS + (ph - 1) * OUTW + (pw - 1)];
  x[i] = (__bf16)v;
}

__global__ void k_cvt_w1(const float* __restrict__ w1, __bf16* __restrict__ wt) {
  long i = (long)blockIdx.x * 256 + threadIdx.x;           // dest: co*4608 + tap*512 + ci
  if (i >= (long)FEAT * KTOT) return;
  int k  = (int)(i % KTOT);
  int co = (int)(i / KTOT);
  int tap = k / FEAT, ci = k % FEAT;
  wt[i] = (__bf16)w1[((long)co * FEAT + ci) * 9 + tap];
}

__global__ void k_cvt_whead(const float* __restrict__ wcls, const float* __restrict__ wbox,
                            __bf16* __restrict__ wh) {
  int i = blockIdx.x * 256 + threadIdx.x;                  // 48*512
  if (i >= 48 * FEAT) return;
  int c = i % FEAT, n = i / FEAT;
  float v = 0.f;
  if (n < 9)        v = wcls[n * FEAT + c];
  else if (n < 45)  v = wbox[(n - 9) * FEAT + c];
  wh[i] = (__bf16)v;
}

__global__ void k_init(float* __restrict__ miou, float* __restrict__ acc) {
  int t = threadIdx.x;
  if (t < 256) miou[t] = 0.f;
  if (t < 8)   acc[t]  = 0.f;
}

// --------------------------------------------------------------- conv (WMMA)
// implicit GEMM per batch: M=2500 (pos), K=4608 (tap*512+ci), N=512 (co).
// Block tile 128x128; 8 waves, each 32x64 (2x4 WMMA 16x16x32 bf16).
// A/B staged via GLOBAL_LOAD_ASYNC_TO_LDS_B128, triple-buffered LDS
// (steady state keeps one 4-load batch in flight: s_wait_asynccnt 4).

__launch_bounds__(256)
__global__ void k_conv(const __bf16* __restrict__ xbp, const __bf16* __restrict__ wt,
                       const float* __restrict__ b1, __bf16* __restrict__ xout) {
  __shared__ __align__(16) unsigned short lA[3][128 * 32];
  __shared__ __align__(16) unsigned short lB[3][128 * 32];

  const int tid    = threadIdx.x;
  const int b      = blockIdx.z;
  const int tile_m = blockIdx.x * 128;
  const int co0    = blockIdx.y * 128;
  const int lane   = tid & 31;
  const int wv     = tid >> 5;
  const int wm     = wv & 3;                 // 4 M groups of 32
  const int wn     = wv >> 2;                // 2 N groups of 64
  const int lr     = lane & 15;
  const int ahof   = (lane < 16) ? 0 : 8;    // A k-half offset (elements)
  const int bhof   = (lane < 16) ? 0 : 8;    // B k-half offset (elements)

  const v8f vzero = {};
  v8f acc[2][4];
#pragma unroll
  for (int i = 0; i < 2; ++i)
#pragma unroll
    for (int j = 0; j < 4; ++j) acc[i][j] = vzero;

  // staging: each thread moves 32B of A and 32B of B per k-step
  const int ar = tid >> 1;                   // 0..127 (A row)
  const int ac = (tid & 1) * 16;             // element offset in row
  const int p  = tile_m + ar;
  const int pm = (p < NPOS) ? p : 0;         // clamp overhang (masked at store)
  const int hh = pm / OUTW, ww = pm % OUTW;
  const int br = tid >> 1;                   // 0..127 (B row)
  const int bc = (tid & 1) * 16;

  // tap-center global base for this thread's A chunk (padded NHWC)
  const __bf16* gA0 = xbp + (((long)b * PADW + hh + 1) * PADW + (ww + 1)) * FEAT + ac;
  const __bf16* gB0 = wt + (long)(co0 + br) * KTOT + bc;
  __builtin_prefetch(gB0, 0, 3);

  auto issue = [&](int kb, int buf) {
    const int tap = kb >> 4;
    const int ci0 = (kb & 15) << 5;
    const int dy = tap / 3 - 1, dx = tap % 3 - 1;
    const __bf16* ga = gA0 + (dy * PADW + dx) * FEAT + ci0;
    const __bf16* gb = gB0 + (tap << 9) + ci0;
    unsigned short* da = &lA[buf][ar * 32 + ac];
    unsigned short* db = &lB[buf][br * 32 + bc];
    async_ld128(da,     ga);
    async_ld128(da + 8, ga + 8);
    async_ld128(db,     gb);
    async_ld128(db + 8, gb + 8);
  };

  auto compute = [&](int buf) {
    v16bf afr[2], bfr[4];
#pragma unroll
    for (int ms = 0; ms < 2; ++ms) {
      const unsigned short* pr = &lA[buf][(wm * 32 + ms * 16 + lr) * 32];
      ((uint4*)&afr[ms])[0] = *(const uint4*)(pr + ahof);
      ((uint4*)&afr[ms])[1] = *(const uint4*)(pr + 16 + ahof);
    }
#pragma unroll
    for (int ns = 0; ns < 4; ++ns) {
      const unsigned short* pr = &lB[buf][(wn * 64 + ns * 16 + lr) * 32];
      ((uint4*)&bfr[ns])[0] = *(const uint4*)(pr + bhof);
      ((uint4*)&bfr[ns])[1] = *(const uint4*)(pr + bhof + 16);
    }
#pragma unroll
    for (int ms = 0; ms < 2; ++ms)
#pragma unroll
      for (int ns = 0; ns < 4; ++ns)
        acc[ms][ns] = __builtin_amdgcn_wmma_f32_16x16x32_bf16(
            false, afr[ms], false, bfr[ns], (short)0, acc[ms][ns], false, false);
  };

  // prologue: two batches in flight
  issue(0, 0);
  issue(1, 1);
  int cur = 0;

  // steady state: batch kb ready (wait<=4 leaves batch kb+1 in flight),
  // then launch batch kb+2 into buffer (kb+2)%3 == (kb-1)%3.
  for (int kb = 0; kb < KB_N - 2; ++kb) {
    wait_async<4>();
    __syncthreads();
    int nb = cur + 2; if (nb >= 3) nb -= 3;
    issue(kb + 2, nb);
    compute(cur);
    cur = (cur + 1 == 3) ? 0 : cur + 1;
  }
  // drain: kb = KB_N-2 (one batch still in flight), then kb = KB_N-1
  wait_async<4>();
  __syncthreads();
  compute(cur);
  cur = (cur + 1 == 3) ? 0 : cur + 1;
  wait_async<0>();
  __syncthreads();
  compute(cur);

  // epilogue: +bias, bf16 NHWC store
#pragma unroll
  for (int ms = 0; ms < 2; ++ms) {
#pragma unroll
    for (int ns = 0; ns < 4; ++ns) {
      const int n = co0 + wn * 64 + ns * 16 + lr;
      const float bias = b1[n];
#pragma unroll
      for (int r = 0; r < 8; ++r) {
        const int m = wm * 32 + ms * 16 + r + ((lane < 16) ? 0 : 8);
        const int pp = tile_m + m;
        if (pp < NPOS)
          xout[((long)b * NPOS + pp) * FEAT + n] = (__bf16)(acc[ms][ns][r] + bias);
      }
    }
  }
}

// ------------------------------------------------------------- heads (WMMA)
// per wave: 16 positions x 48 outputs (9 cls + 36 box + 3 pad), K=512

__launch_bounds__(128)
__global__ void k_head(const __bf16* __restrict__ xout, const __bf16* __restrict__ wh,
                       const float* __restrict__ bcls, const float* __restrict__ bbox,
                       float* __restrict__ logits, float* __restrict__ pred) {
  const int b    = blockIdx.y;
  const int tile = blockIdx.x * 4 + (threadIdx.x >> 5);
  if (tile >= 157) return;
  const int lane = threadIdx.x & 31;
  const int lr   = lane & 15;
  const int ahof = (lane < 16) ? 0 : 8;
  const int bhof = (lane < 16) ? 0 : 8;

  const v8f vzero = {};
  v8f acc[3] = {vzero, vzero, vzero};

  const int m  = tile * 16 + lr;
  const int pm = (m < NPOS) ? m : 0;
  const __bf16* arow = xout + ((long)b * NPOS + pm) * FEAT;

  for (int k0 = 0; k0 < FEAT; k0 += 32) {
    v16bf af;
    ((uint4*)&af)[0] = *(const uint4*)(arow + k0 + ahof);
    ((uint4*)&af)[1] = *(const uint4*)(arow + k0 + 16 + ahof);
#pragma unroll
    for (int nt = 0; nt < 3; ++nt) {
      v16bf bf;
      const __bf16* brow = wh + (long)(nt * 16 + lr) * FEAT + k0 + bhof;
      ((uint4*)&bf)[0] = *(const uint4*)(brow);
      ((uint4*)&bf)[1] = *(const uint4*)(brow + 16);
      acc[nt] = __builtin_amdgcn_wmma_f32_16x16x32_bf16(
          false, af, false, bf, (short)0, acc[nt], false, false);
    }
  }

#pragma unroll
  for (int nt = 0; nt < 3; ++nt) {
    const int n = nt * 16 + lr;
#pragma unroll
    for (int r = 0; r < 8; ++r) {
      const int mo = tile * 16 + r + ((lane < 16) ? 0 : 8);
      if (mo < NPOS && n < 45) {
        if (n < 9) {
          logits[(long)b * NANCH + mo * AA + n] = acc[nt][r] + bcls[n];
        } else {
          const int d = (n - 9) & 3, a = (n - 9) >> 2;
          pred[(((long)b * NANCH + mo * AA + a) << 2) + d] = acc[nt][r] + bbox[n - 9];
        }
      }
    }
  }
}

// ---------------------------------------------------------------- loss kernels

__device__ __forceinline__ void anchor_of(int n, float& x1, float& y1, float& x2, float& y2) {
  const int a = n % AA, p = n / AA;
  const int j = p % OUTW, i = p / OUTW;
  const float S[3] = {2.f, 4.f, 6.f};
  const float R[3] = {0.5f, 1.f, 1.5f};
  const float s = S[a % 3], r = R[a / 3];
  const float w = s * r, h = s;
  const float cx = (float)i + 0.5f, cy = (float)j + 0.5f;
  x1 = fminf(fmaxf(cx - 0.5f * w, 0.f), 50.f);
  x2 = fminf(fmaxf(cx + 0.5f * w, 0.f), 50.f);
  y1 = fminf(fmaxf(cy - 0.5f * h, 0.f), 50.f);
  y2 = fminf(fmaxf(cy + 0.5f * h, 0.f), 50.f);
}

__device__ __forceinline__ float iou_one(float ax1, float ay1, float ax2, float ay2,
                                         float area_a,
                                         float gx1, float gy1, float gx2, float gy2) {
  const float ix1 = fmaxf(ax1, gx1), iy1 = fmaxf(ay1, gy1);
  const float ix2 = fminf(ax2, gx2), iy2 = fminf(ay2, gy2);
  const float iw = fmaxf(ix2 - ix1, 0.f), ih = fmaxf(iy2 - iy1, 0.f);
  const float inter = iw * ih;
  const float area_g = (gx2 - gx1) * (gy2 - gy1);
  return inter / (area_a + area_g - inter + 1e-8f);
}

__device__ __forceinline__ float sl1(float d) {
  const float ad = fabsf(d);
  return (ad < 1.f) ? 0.5f * d * d : ad - 0.5f;
}
__device__ __forceinline__ float softplus(float x) {
  const float ax = fabsf(x);
  return fmaxf(x, 0.f) + log1pf(__expf(-ax));
}

__launch_bounds__(256)
__global__ void k_ioumax(const float* __restrict__ gt, float* __restrict__ miou) {
  __shared__ float red[256];
  const int b = blockIdx.y;
  const int n = blockIdx.x * 256 + threadIdx.x;
  const bool valid = (n < NANCH);
  float ax1, ay1, ax2, ay2;
  anchor_of(valid ? n : 0, ax1, ay1, ax2, ay2);
  const float area_a = (ax2 - ax1) * (ay2 - ay1);
  for (int m = 0; m < 16; ++m) {
    const float* g = gt + ((long)b * 16 + m) * 4;
    const float inv = 1.f / 16.f;
    const float iou = iou_one(ax1, ay1, ax2, ay2, area_a,
                              g[0] * inv, g[1] * inv, g[2] * inv, g[3] * inv);
    red[threadIdx.x] = valid ? iou : 0.f;
    __syncthreads();
    for (int s = 128; s > 0; s >>= 1) {
      if (threadIdx.x < s) red[threadIdx.x] = fmaxf(red[threadIdx.x], red[threadIdx.x + s]);
      __syncthreads();
    }
    if (threadIdx.x == 0)
      atomicMax((int*)&miou[b * 16 + m], __float_as_int(red[0]));
    __syncthreads();
  }
}

__launch_bounds__(256)
__global__ void k_loss(const float* __restrict__ gt, const float* __restrict__ miou,
                       const float* __restrict__ logits, const float* __restrict__ pred,
                       float* __restrict__ accum, float* __restrict__ out) {
  __shared__ float red[256];
  const int b = blockIdx.y;
  const int n = blockIdx.x * 256 + threadIdx.x;
  const bool valid = (n < NANCH);

  float v_npos = 0.f, v_reg = 0.f, v_pos = 0.f, v_negc = 0.f, v_negl = 0.f;

  if (valid) {
    float ax1, ay1, ax2, ay2;
    anchor_of(n, ax1, ay1, ax2, ay2);
    const float area_a = (ax2 - ax1) * (ay2 - ay1);
    const float acx = 0.5f * (ax1 + ax2), acy = 0.5f * (ay1 + ay2);
    const float aw = fmaxf(ax2 - ax1, 1e-6f), ah = fmaxf(ay2 - ay1, 1e-6f);

    const float l = logits[(long)b * NANCH + n];
    const float* pp = pred + (((long)b * NANCH + n) << 2);
    const float p0 = pp[0], p1 = pp[1], p2 = pp[2], p3 = pp[3];

    bool allneg = true;
    for (int m = 0; m < 16; ++m) {
      const float* g = gt + ((long)b * 16 + m) * 4;
      const float inv = 1.f / 16.f;
      const float gx1 = g[0] * inv, gy1 = g[1] * inv, gx2 = g[2] * inv, gy2 = g[3] * inv;
      const float iou = iou_one(ax1, ay1, ax2, ay2, area_a, gx1, gy1, gx2, gy2);
      const float mx = miou[b * 16 + m];
      const bool pos = ((iou == mx) && (mx > 0.f)) || (iou > 0.7f);
      if (iou >= 0.3f) allneg = false;
      if (pos) {
        v_npos += 1.f;
        const float gcx = 0.5f * (gx1 + gx2), gcy = 0.5f * (gy1 + gy2);
        const float gw = fmaxf(gx2 - gx1, 1e-6f), gh = fmaxf(gy2 - gy1, 1e-6f);
        const float tx = (gcx - acx) / aw, ty = (gcy - acy) / ah;
        const float tw = __logf(gw / aw), th = __logf(gh / ah);
        v_reg += sl1(tx - p0) + sl1(ty - p1) + sl1(tw - p2) + sl1(th - p3);
      }
    }
    v_pos = v_npos * softplus(-l);
    if (allneg) { v_negc = 1.f; v_negl = softplus(l); }

    const float pcx = acx + p0 * aw, pcy = acy + p1 * ah;
    const float pw = aw * __expf(p2), ph = ah * __expf(p3);
    float* o = out + 1 + (((long)b * NANCH + n) << 2);
    o[0] = pcx - 0.5f * pw;
    o[1] = pcy - 0.5f * ph;
    o[2] = pcx + 0.5f * pw;
    o[3] = pcy + 0.5f * ph;
  }

  float vals[5] = {v_npos, v_reg, v_pos, v_negc, v_negl};
#pragma unroll
  for (int q = 0; q < 5; ++q) {
    red[threadIdx.x] = vals[q];
    __syncthreads();
    for (int s = 128; s > 0; s >>= 1) {
      if (threadIdx.x < s) red[threadIdx.x] += red[threadIdx.x + s];
      __syncthreads();
    }
    if (threadIdx.x == 0) atomicAdd(&accum[q], red[0]);
    __syncthreads();
  }
}

__global__ void k_final(const float* __restrict__ accum, float* __restrict__ out) {
  if (threadIdx.x == 0 && blockIdx.x == 0) {
    const float npos = fmaxf(accum[0], 1.f);
    const float regs = accum[1];
    const float poss = accum[2];
    const float nneg = fmaxf(accum[3], 1.f);
    const float negl = accum[4];
    const float cls = 0.5f * (poss / npos + negl / nneg);
    out[0] = 1.0f * cls + 5.0f * (regs / (4.f * npos));
  }
}

// ---------------------------------------------------------------- launcher

extern "C" void kernel_launch(void* const* d_in, const int* in_sizes, int n_in,
                              void* d_out, int out_size, void* d_ws, size_t ws_size,
                              hipStream_t stream) {
  const float* features = (const float*)d_in[0];
  const float* gt       = (const float*)d_in[1];
  const float* w1       = (const float*)d_in[2];
  const float* b1       = (const float*)d_in[3];
  const float* wcls     = (const float*)d_in[4];
  const float* bcls     = (const float*)d_in[5];
  const float* wbox     = (const float*)d_in[6];
  const float* bbox     = (const float*)d_in[7];
  float* out = (float*)d_out;
  char*  ws  = (char*)d_ws;

  size_t off = 0;
  auto take = [&](size_t bytes) {
    size_t o = off;
    off = (off + bytes + 255) & ~(size_t)255;
    return o;
  };
  const size_t o_xb = take((size_t)BB * PADW * PADW * FEAT * 2);  // padded NHWC bf16
  const size_t o_wt = take((size_t)FEAT * KTOT * 2);              // conv weights bf16
  const size_t o_wh = take((size_t)48 * FEAT * 2);                // head weights bf16
  const size_t o_xo = take((size_t)BB * NPOS * FEAT * 2);         // conv output bf16
  const size_t o_lg = take((size_t)BB * NANCH * 4);               // logits f32
  const size_t o_pr = take((size_t)BB * NANCH * 4 * 4);           // pred_off f32
  const size_t o_mi = take((size_t)256 * 4);                      // max iou per (b,m)
  const size_t o_ac = take((size_t)8 * 4);                        // loss accumulators

  __bf16* xb = (__bf16*)(ws + o_xb);
  __bf16* wt = (__bf16*)(ws + o_wt);
  __bf16* wh = (__bf16*)(ws + o_wh);
  __bf16* xo = (__bf16*)(ws + o_xo);
  float*  lg = (float*)(ws + o_lg);
  float*  pr = (float*)(ws + o_pr);
  float*  mi = (float*)(ws + o_mi);
  float*  ac = (float*)(ws + o_ac);

  {
    long tot = (long)BB * PADW * PADW * FEAT;
    k_cvt_feat_pad<<<dim3((unsigned)((tot + 255) / 256)), dim3(256), 0, stream>>>(features, xb);
  }
  {
    long tot = (long)FEAT * KTOT;
    k_cvt_w1<<<dim3((unsigned)((tot + 255) / 256)), dim3(256), 0, stream>>>(w1, wt);
  }
  k_cvt_whead<<<dim3(96), dim3(256), 0, stream>>>(wcls, wbox, wh);
  k_init<<<dim3(1), dim3(256), 0, stream>>>(mi, ac);

  k_conv<<<dim3(20, 4, 16), dim3(256), 0, stream>>>(xb, wt, b1, xo);
  k_head<<<dim3(40, 16), dim3(128), 0, stream>>>(xo, wh, bcls, bbox, lg, pr);

  k_ioumax<<<dim3(88, 16), dim3(256), 0, stream>>>(gt, mi);
  k_loss<<<dim3(88, 16), dim3(256), 0, stream>>>(gt, mi, lg, pr, ac, out);
  k_final<<<dim3(1), dim3(1), 0, stream>>>(ac, out);
}
